// GraphTransformerLayer_41059887350105
// MI455X (gfx1250) — matrix operand order repaired and checked by reading, not compiled, not run on hardware
//
#include <hip/hip_runtime.h>
#include <hip/hip_bf16.h>

#define N_TOK 3072
#define DMODEL 256
#define NHEAD 8
#define HD (NHEAD * DMODEL) /* 2048 */
#define EPSV 1e-5f

typedef __attribute__((ext_vector_type(16))) _Float16 v16h;
typedef __attribute__((ext_vector_type(8)))  _Float16 v8h;
typedef __attribute__((ext_vector_type(8)))  float    v8f;

__device__ __forceinline__ v8f wmma16(v16h a, v16h b, v8f c) {
  // D = A(16x32,f16) * B(32x16,f16) + C(16x16,f32)
  return __builtin_amdgcn_wmma_f32_16x16x32_f16(
      /*neg_a*/false, a, /*neg_b*/false, b,
      /*c_mod*/(short)0, c, /*reuse_a*/false, /*reuse_b*/false);
}

// A operand: row-major (16 rows x >=32 cols, leading dim ld halves).
// ISA layout: lanes 0-15 hold K 0-7 & 16-23, lanes 16-31 hold K 8-15 & 24-31.
__device__ __forceinline__ v16h load_a_tile(const _Float16* p, int ld) {
  int lane = threadIdx.x & 31;
  const _Float16* q = p + (lane & 15) * ld + ((lane >> 4) << 3);
  v16h r;
  *(v8h*)&r       = *(const v8h*)q;
  *((v8h*)&r + 1) = *(const v8h*)(q + 16);
  return r;
}

// B operand loaded from Bt (row-major, rows = output cols, ld halves along K).
// ISA layout: lanes 0-15 hold K 0-7 & 16-23, lanes 16-31 hold K 8-15 & 24-31.
__device__ __forceinline__ v16h load_b_tile(const _Float16* p, int ld) {
  int lane = threadIdx.x & 31;
  const _Float16* q = p + (lane & 15) * ld + ((lane >> 4) << 3);
  v16h r;
  *(v8h*)&r       = *(const v8h*)q;
  *((v8h*)&r + 1) = *(const v8h*)(q + 16);
  return r;
}

// ---------- small prep kernels ----------
__global__ void k_cvt(const float* __restrict__ s, _Float16* __restrict__ d, int n) {
  int i = blockIdx.x * 256 + threadIdx.x;
  if (i < n) d[i] = (_Float16)s[i];
}

// src: batch of R x C f32 row-major -> dst: batch of C x R f16 (transposed)
__global__ void k_transpose(const float* __restrict__ src, _Float16* __restrict__ dst,
                            int R, int C) {
  size_t base = (size_t)blockIdx.y * (size_t)R * (size_t)C;
  int idx = blockIdx.x * 256 + threadIdx.x;
  if (idx < R * C) {
    int r = idx / C, c = idx % C;
    dst[base + (size_t)c * R + r] = (_Float16)src[base + idx];
  }
}

// ---------- QKV projections ----------
// grid.x: 384 (8 waves/block * 384 = 3072 tiles = 192 mtiles x 16 ntiles)
// grid.y: 24 = {Q,K,V} x 8 heads
__global__ void k_qkv(const _Float16* __restrict__ Xh,
                      const _Float16* __restrict__ WQt,
                      const _Float16* __restrict__ WKt,
                      const _Float16* __restrict__ WVt,
                      _Float16* __restrict__ Qb, _Float16* __restrict__ Kb,
                      _Float16* __restrict__ Vt) {
  int wid = threadIdx.x >> 5;
  int tile = blockIdx.x * 8 + wid;
  int mt = tile >> 4, nt = tile & 15;
  int z = blockIdx.y, kind = z >> 3, h = z & 7;
  const _Float16* Bt = (kind == 0 ? WQt : (kind == 1 ? WKt : WVt)) + h * DMODEL * DMODEL;
  v8f acc = {};
#pragma unroll
  for (int kt = 0; kt < 8; ++kt) {
    v16h a = load_a_tile(Xh + mt * 16 * DMODEL + kt * 32, DMODEL);
    v16h b = load_b_tile(Bt + nt * 16 * DMODEL + kt * 32, DMODEL);
    acc = wmma16(a, b, acc);
  }
  int lane = threadIdx.x & 31, col = lane & 15, rb = (lane >> 4) << 3;
  if (kind < 2) {
    _Float16* dst = (kind == 0 ? Qb : Kb) + h * N_TOK * DMODEL;
#pragma unroll
    for (int i = 0; i < 8; ++i)
      dst[(mt * 16 + rb + i) * DMODEL + nt * 16 + col] = (_Float16)acc[i];
  } else {
    _Float16* dst = Vt + h * DMODEL * N_TOK;   // transposed: [e][n]
#pragma unroll
    for (int i = 0; i < 8; ++i)
      dst[(nt * 16 + col) * N_TOK + mt * 16 + rb + i] = (_Float16)acc[i];
  }
}

// ---------- flash-style attention ----------
// block = 256 threads (8 waves); each wave handles 16 rows of one head,
// iterating the key/value dimension in 64-column chunks.
// grid: (N/128, H)
__global__ void k_attn(const _Float16* __restrict__ Qb,
                       const _Float16* __restrict__ Kb,
                       const _Float16* __restrict__ Vt,
                       _Float16* __restrict__ AttnCat) {
  __shared__ __align__(16) _Float16 lp[8][16 * 64];
  int wid = threadIdx.x >> 5;
  int lane = threadIdx.x & 31, col = lane & 15, rb = (lane >> 4) << 3;
  int h = blockIdx.y;
  int r0 = blockIdx.x * 128 + wid * 16;
  const _Float16* Qh = Qb + h * N_TOK * DMODEL;
  const _Float16* Kh = Kb + h * N_TOK * DMODEL;
  const _Float16* Vh = Vt + h * DMODEL * N_TOK;
  _Float16* myl = lp[wid];

  v8f acc[16] = {};
  float rm[8], rs[8];
#pragma unroll
  for (int i = 0; i < 8; ++i) { rm[i] = -3.0e38f; rs[i] = 0.f; }

  for (int m0 = 0; m0 < N_TOK; m0 += 64) {
    // ---- scores: 16x64 tile = 4 C-tiles, K-loop over D=256 ----
    v8f s[4] = {};
#pragma unroll
    for (int kt = 0; kt < 8; ++kt) {
      v16h a = load_a_tile(Qh + r0 * DMODEL + kt * 32, DMODEL);
#pragma unroll
      for (int j = 0; j < 4; ++j) {
        v16h b = load_b_tile(Kh + (m0 + j * 16) * DMODEL + kt * 32, DMODEL);
        s[j] = wmma16(a, b, s[j]);
      }
    }
    // ---- prefetch next chunk's K (64 rows x 512B) and V (256 rows x 128B) ----
    if (m0 + 64 < N_TOK) {
      const char* pk = (const char*)(Kh + (m0 + 64) * DMODEL);
#pragma unroll
      for (int j = 0; j < 8; ++j)
        __builtin_prefetch(pk + lane * 128 + j * 4096, 0, 1);
      const char* pv = (const char*)(Vh + m0 + 64);
#pragma unroll
      for (int j = 0; j < 8; ++j)
        __builtin_prefetch(pv + (size_t)(lane + j * 32) * (N_TOK * 2), 0, 1);
    }
    // ---- online softmax over the 64 columns ----
    float corr[8];
#pragma unroll
    for (int i = 0; i < 8; ++i) {
      int row = r0 + rb + i;
      float v[4];
#pragma unroll
      for (int j = 0; j < 4; ++j) {
        v[j] = s[j][i] * 0.0625f;                    // / sqrt(256)
        if (row == m0 + j * 16 + col) v[j] = -3.0e38f; // diagonal masked out
      }
      float cm = fmaxf(fmaxf(v[0], v[1]), fmaxf(v[2], v[3]));
      cm = fmaxf(cm, __shfl_xor(cm, 1, 32));
      cm = fmaxf(cm, __shfl_xor(cm, 2, 32));
      cm = fmaxf(cm, __shfl_xor(cm, 4, 32));
      cm = fmaxf(cm, __shfl_xor(cm, 8, 32));
      float nm = fmaxf(rm[i], cm);
      float c  = __expf(rm[i] - nm);
      float p[4], ps = 0.f;
#pragma unroll
      for (int j = 0; j < 4; ++j) { p[j] = __expf(v[j] - nm); ps += p[j]; }
      ps += __shfl_xor(ps, 1, 32);
      ps += __shfl_xor(ps, 2, 32);
      ps += __shfl_xor(ps, 4, 32);
      ps += __shfl_xor(ps, 8, 32);
      rs[i] = rs[i] * c + ps;
      rm[i] = nm;
      corr[i] = c;
#pragma unroll
      for (int j = 0; j < 4; ++j)
        myl[(rb + i) * 64 + j * 16 + col] = (_Float16)p[j];
    }
#pragma unroll
    for (int t = 0; t < 16; ++t)
#pragma unroll
      for (int i = 0; i < 8; ++i)
        acc[t][i] *= corr[i];
    asm volatile("s_wait_dscnt 0" ::: "memory");
    v16h pa0 = load_a_tile(myl, 64);        // P columns m0..m0+31 as A operand
    v16h pa1 = load_a_tile(myl + 32, 64);   // P columns m0+32..m0+63
#pragma unroll
    for (int t = 0; t < 16; ++t) {
      v16h vb0 = load_b_tile(Vh + (t * 16) * N_TOK + m0, N_TOK);
      acc[t] = wmma16(pa0, vb0, acc[t]);
      v16h vb1 = load_b_tile(Vh + (t * 16) * N_TOK + m0 + 32, N_TOK);
      acc[t] = wmma16(pa1, vb1, acc[t]);
    }
  }
  // ---- epilogue: 8 reciprocals + 128 muls (not 128 divides) ----
  float rinv[8];
#pragma unroll
  for (int i = 0; i < 8; ++i) rinv[i] = 1.0f / rs[i];
#pragma unroll
  for (int t = 0; t < 16; ++t) {
#pragma unroll
    for (int i = 0; i < 8; ++i) {
      float o = acc[t][i] * rinv[i];
      AttnCat[(size_t)(r0 + rb + i) * HD + h * DMODEL + t * 16 + col] = (_Float16)o;
    }
  }
}

// ---------- attn projection: (N x 2048) @ (2048 x 256), fused (+b)/sqrt(deg) ----------
__global__ void k_ao(const _Float16* __restrict__ AttnCat,
                     const _Float16* __restrict__ LAWt,
                     const float* __restrict__ lab,
                     const float* __restrict__ deg,
                     _Float16* __restrict__ AOs) {
  int wid = threadIdx.x >> 5;
  int tile = blockIdx.x * 8 + wid;
  int mt = tile >> 4, nt = tile & 15;
  v8f acc = {};
#pragma unroll 8
  for (int kt = 0; kt < 64; ++kt) {
    v16h a = load_a_tile(AttnCat + (size_t)mt * 16 * HD + kt * 32, HD);
    v16h b = load_b_tile(LAWt + (size_t)nt * 16 * HD + kt * 32, HD);
    acc = wmma16(a, b, acc);
  }
  int lane = threadIdx.x & 31, col = lane & 15, rb = (lane >> 4) << 3;
  float bv = lab[nt * 16 + col];
#pragma unroll
  for (int i = 0; i < 8; ++i) {
    int n = mt * 16 + rb + i;
    float o = (acc[i] + bv) * rsqrtf(deg[n]);
    AOs[n * DMODEL + nt * 16 + col] = (_Float16)o;
  }
}

// ---------- fused concat-GEMM: X1pre = Xh@W1a + AOs@W1b + lin_b ----------
__global__ void k_x1(const _Float16* __restrict__ Xh,
                     const _Float16* __restrict__ AOs,
                     const _Float16* __restrict__ LWt,   // 256 x 512 (out_e x k)
                     const float* __restrict__ lb,
                     float* __restrict__ X1pre) {
  int wid = threadIdx.x >> 5;
  int tile = blockIdx.x * 8 + wid;
  int mt = tile >> 4, nt = tile & 15;
  v8f acc = {};
#pragma unroll
  for (int kt = 0; kt < 8; ++kt)
    acc = wmma16(load_a_tile(Xh + mt * 16 * DMODEL + kt * 32, DMODEL),
                 load_b_tile(LWt + nt * 16 * 512 + kt * 32, 512), acc);
#pragma unroll
  for (int kt = 0; kt < 8; ++kt)
    acc = wmma16(load_a_tile(AOs + mt * 16 * DMODEL + kt * 32, DMODEL),
                 load_b_tile(LWt + nt * 16 * 512 + 256 + kt * 32, 512), acc);
  int lane = threadIdx.x & 31, col = lane & 15, rb = (lane >> 4) << 3;
  float bv = lb[nt * 16 + col];
#pragma unroll
  for (int i = 0; i < 8; ++i)
    X1pre[(mt * 16 + rb + i) * DMODEL + nt * 16 + col] = acc[i] + bv;
}

// ---------- FFN ----------
__global__ void k_fnn1(const _Float16* __restrict__ X1h, const _Float16* __restrict__ F1t,
                       const float* __restrict__ b1, _Float16* __restrict__ X2a) {
  int wid = threadIdx.x >> 5;
  int tile = blockIdx.x * 8 + wid;
  int mt = tile >> 4, nt = tile & 15;
  v8f acc = {};
#pragma unroll
  for (int kt = 0; kt < 8; ++kt)
    acc = wmma16(load_a_tile(X1h + mt * 16 * DMODEL + kt * 32, DMODEL),
                 load_b_tile(F1t + nt * 16 * DMODEL + kt * 32, DMODEL), acc);
  int lane = threadIdx.x & 31, col = lane & 15, rb = (lane >> 4) << 3;
  float bv = b1[nt * 16 + col];
#pragma unroll
  for (int i = 0; i < 8; ++i) {
    float o = fmaxf(acc[i] + bv, 0.f);          // relu
    X2a[(mt * 16 + rb + i) * DMODEL + nt * 16 + col] = (_Float16)o;
  }
}

__global__ void k_fnn2(const _Float16* __restrict__ X2a, const _Float16* __restrict__ F2t,
                       const float* __restrict__ b2, float* __restrict__ X2pre) {
  int wid = threadIdx.x >> 5;
  int tile = blockIdx.x * 8 + wid;
  int mt = tile >> 4, nt = tile & 15;
  v8f acc = {};
#pragma unroll
  for (int kt = 0; kt < 8; ++kt)
    acc = wmma16(load_a_tile(X2a + mt * 16 * DMODEL + kt * 32, DMODEL),
                 load_b_tile(F2t + nt * 16 * DMODEL + kt * 32, DMODEL), acc);
  int lane = threadIdx.x & 31, col = lane & 15, rb = (lane >> 4) << 3;
  float bv = b2[nt * 16 + col];
#pragma unroll
  for (int i = 0; i < 8; ++i)
    X2pre[(mt * 16 + rb + i) * DMODEL + nt * 16 + col] = acc[i] + bv;
}

// ---------- residual + LayerNorm (wave per row) ----------
__global__ void k_ln(const float* __restrict__ res, const float* __restrict__ y,
                     const float* __restrict__ g, const float* __restrict__ b,
                     float* __restrict__ outf, _Float16* __restrict__ outh) {
  int wid = threadIdx.x >> 5, lane = threadIdx.x & 31;
  int row = blockIdx.x * 8 + wid;
  const float* pr = res + row * DMODEL;
  const float* py = y + row * DMODEL;
  float v[8], s = 0.f;
#pragma unroll
  for (int j = 0; j < 8; ++j) { v[j] = pr[lane + j * 32] + py[lane + j * 32]; s += v[j]; }
#pragma unroll
  for (int m = 1; m < 32; m <<= 1) s += __shfl_xor(s, m, 32);
  float mu = s * (1.f / DMODEL);
  float q = 0.f;
#pragma unroll
  for (int j = 0; j < 8; ++j) { float d = v[j] - mu; q += d * d; }
#pragma unroll
  for (int m = 1; m < 32; m <<= 1) q += __shfl_xor(q, m, 32);
  float inv = rsqrtf(q * (1.f / DMODEL) + EPSV);
#pragma unroll
  for (int j = 0; j < 8; ++j) {
    int c = lane + j * 32;
    float o = (v[j] - mu) * inv * g[c] + b[c];
    outf[row * DMODEL + c] = o;
    if (outh) outh[row * DMODEL + c] = (_Float16)o;
  }
}

extern "C" void kernel_launch(void* const* d_in, const int* in_sizes, int n_in,
                              void* d_out, int out_size, void* d_ws, size_t ws_size,
                              hipStream_t stream) {
  (void)in_sizes; (void)n_in; (void)out_size; (void)ws_size;
  const float* input_x = (const float*)d_in[0];
  const float* deg     = (const float*)d_in[1];
  const float* WQ      = (const float*)d_in[2];
  const float* WK      = (const float*)d_in[3];
  const float* WV      = (const float*)d_in[4];
  const float* law     = (const float*)d_in[5];
  const float* lab     = (const float*)d_in[6];
  const float* lw      = (const float*)d_in[7];
  const float* lb      = (const float*)d_in[8];
  const float* ln1g    = (const float*)d_in[9];
  const float* ln1b    = (const float*)d_in[10];
  const float* f1w     = (const float*)d_in[11];
  const float* f1b     = (const float*)d_in[12];
  const float* f2w     = (const float*)d_in[13];
  const float* f2b     = (const float*)d_in[14];
  const float* ln2g    = (const float*)d_in[15];
  const float* ln2b    = (const float*)d_in[16];
  float* out = (float*)d_out;

  char* ws = (char*)d_ws;
  size_t off = 0;
  auto alloc = [&](size_t bytes) { char* p = ws + off; off += (bytes + 255) & ~(size_t)255; return p; };
  _Float16* Xh    = (_Float16*)alloc((size_t)N_TOK * DMODEL * 2);
  _Float16* WQt   = (_Float16*)alloc((size_t)NHEAD * DMODEL * DMODEL * 2);
  _Float16* WKt   = (_Float16*)alloc((size_t)NHEAD * DMODEL * DMODEL * 2);
  _Float16* WVt   = (_Float16*)alloc((size_t)NHEAD * DMODEL * DMODEL * 2);
  _Float16* LAWt  = (_Float16*)alloc((size_t)DMODEL * HD * 2);
  _Float16* LWt   = (_Float16*)alloc((size_t)DMODEL * 2 * DMODEL * 2);
  _Float16* F1t   = (_Float16*)alloc((size_t)DMODEL * DMODEL * 2);
  _Float16* F2t   = (_Float16*)alloc((size_t)DMODEL * DMODEL * 2);
  _Float16* Qb    = (_Float16*)alloc((size_t)NHEAD * N_TOK * DMODEL * 2);
  _Float16* Kb    = (_Float16*)alloc((size_t)NHEAD * N_TOK * DMODEL * 2);
  _Float16* Vt    = (_Float16*)alloc((size_t)NHEAD * DMODEL * N_TOK * 2);
  _Float16* ACat  = (_Float16*)alloc((size_t)N_TOK * HD * 2);
  _Float16* AOs   = (_Float16*)alloc((size_t)N_TOK * DMODEL * 2);
  float*    X1pre = (float*)   alloc((size_t)N_TOK * DMODEL * 4);
  float*    X1f   = (float*)   alloc((size_t)N_TOK * DMODEL * 4);
  _Float16* X1h   = (_Float16*)alloc((size_t)N_TOK * DMODEL * 2);
  _Float16* X2a   = (_Float16*)alloc((size_t)N_TOK * DMODEL * 2);
  float*    X2pre = (float*)   alloc((size_t)N_TOK * DMODEL * 4);

  // prep: convert x, transpose+convert all weights
  k_cvt<<<dim3((N_TOK * DMODEL + 255) / 256), dim3(256), 0, stream>>>(input_x, Xh, N_TOK * DMODEL);
  k_transpose<<<dim3(256, NHEAD), dim3(256), 0, stream>>>(WQ, WQt, DMODEL, DMODEL);
  k_transpose<<<dim3(256, NHEAD), dim3(256), 0, stream>>>(WK, WKt, DMODEL, DMODEL);
  k_transpose<<<dim3(256, NHEAD), dim3(256), 0, stream>>>(WV, WVt, DMODEL, DMODEL);
  k_transpose<<<dim3(2048, 1), dim3(256), 0, stream>>>(law, LAWt, HD, DMODEL);
  k_transpose<<<dim3(512, 1), dim3(256), 0, stream>>>(lw, LWt, 2 * DMODEL, DMODEL);
  k_transpose<<<dim3(256, 1), dim3(256), 0, stream>>>(f1w, F1t, DMODEL, DMODEL);
  k_transpose<<<dim3(256, 1), dim3(256), 0, stream>>>(f2w, F2t, DMODEL, DMODEL);

  // QKV
  k_qkv<<<dim3(384, 24), dim3(256), 0, stream>>>(Xh, WQt, WKt, WVt, Qb, Kb, Vt);
  // attention (writes attn_cat layout directly)
  k_attn<<<dim3(N_TOK / 128, NHEAD), dim3(256), 0, stream>>>(Qb, Kb, Vt, ACat);
  // attn projection + /sqrt(deg)
  k_ao<<<dim3(384), dim3(256), 0, stream>>>(ACat, LAWt, lab, deg, AOs);
  // fused concat linear
  k_x1<<<dim3(384), dim3(256), 0, stream>>>(Xh, AOs, LWt, lb, X1pre);
  // LN1 (residual with input_x); emits f32 x1 and f16 x1
  k_ln<<<dim3(384), dim3(256), 0, stream>>>(input_x, X1pre, ln1g, ln1b, X1f, X1h);
  // FFN
  k_fnn1<<<dim3(384), dim3(256), 0, stream>>>(X1h, F1t, f1b, X2a);
  k_fnn2<<<dim3(384), dim3(256), 0, stream>>>(X2a, F2t, f2b, X2pre);
  // LN2 (residual with x1) -> final output
  k_ln<<<dim3(384), dim3(256), 0, stream>>>(X1f, X2pre, ln2g, ln2b, out, (_Float16*)nullptr);
}
